// MHeadOA_14628658610725
// MI455X (gfx1250) — matrix operand order repaired and checked by reading, not compiled, not run on hardware
//
#include <hip/hip_runtime.h>
#include <hip/hip_bf16.h>

#define BB 4
#define CC 256
#define NN 2048
#define HH 8
#define DD 64
#define HC (HH * CC)

typedef __attribute__((ext_vector_type(16))) __bf16 v16bf;
typedef __attribute__((ext_vector_type(8)))  float  v8f;

union FragU { uint4 q[2]; v16bf v; };

// A fragment chunks at immediate element offsets from a per-lane base pointer.
// Base = A + (row0 + lane&15)*lda + (lane&16 ? 8 : 0).
// ISA A 16x32 layout: lanes 0-15 K {0..7,16..23}; lanes 16-31 K {8..15,24..31}.
__device__ __forceinline__ v16bf frag_a_at(const __bf16* p, int koff) {
  FragU f;
  f.q[0] = *(const uint4*)(p + koff);        // K kb..kb+7
  f.q[1] = *(const uint4*)(p + koff + 16);   // K kb+16..kb+23
  return f.v;
}

// B fragment from transposed storage Bt[N][K].
// Base = Bt + (col0 + lane&15)*ldb + (lane&16 ? 8 : 0).
// Lanes 0-15 hold K {0..7,16..23}, lanes 16-31 K {8..15,24..31} (same as A).
__device__ __forceinline__ v16bf frag_b_at(const __bf16* p, int koff) {
  FragU f;
  f.q[0] = *(const uint4*)(p + koff);        // K kb..kb+7
  f.q[1] = *(const uint4*)(p + koff + 16);   // K kb+16..kb+23
  return f.v;
}

enum { EPI_BF16 = 0, EPI_F32, EPI_BIAS_BF16, EPI_SUBX_T_BF16, EPI_HEAD, EPI_FINAL };

// C[i][j] = sum_k A[i][k] * Bt[j][k].
// 8 waves/block (2 along i, 4 along j); each wave computes a (16*WI)x(16*WJ) tile.
// K-loop unrolled by 2 with ping-pong fragment buffers; fragment base pointers are
// hoisted and bumped by a uniform stride so all loads use immediate offsets.
// Requires K % 64 == 0. Block tile = (32*WI) x (64*WJ).
// grid.x = Nd/(64*WJ), grid.y = M/(32*WI), grid.z = batch.
template <int MODE, int WI, int WJ>
__global__ __launch_bounds__(256) void wmma_gemm(
    const __bf16* __restrict__ A, const __bf16* __restrict__ Bt, int K, int lda, int ldb,
    long long batchA, long long batchB,
    float* __restrict__ outF, __bf16* __restrict__ outB, int ldo, long long batchO,
    const float* __restrict__ p0, long long p0b, int ldp,
    const float* __restrict__ p1, const float* __restrict__ p2, const float* __restrict__ p3,
    const float* __restrict__ p4, const float* __restrict__ p5, int coff) {
  const int lane  = threadIdx.x & 31;
  const int wave  = threadIdx.x >> 5;
  const int z     = blockIdx.z;
  const int tileI = (blockIdx.y * 2 + (wave >> 2)) * (16 * WI);
  const int tileJ = (blockIdx.x * 4 + (wave & 3)) * (16 * WJ);

  // per-lane fragment base pointers, hoisted out of the K loop
  const __bf16* pa[WI];
  const __bf16* pb[WJ];
#pragma unroll
  for (int i = 0; i < WI; ++i)
    pa[i] = A + (size_t)z * batchA +
            (size_t)(tileI + 16 * i + (lane & 15)) * lda + ((lane & 16) ? 8 : 0);
#pragma unroll
  for (int j = 0; j < WJ; ++j)
    pb[j] = Bt + (size_t)z * batchB +
            (size_t)(tileJ + 16 * j + (lane & 15)) * ldb + ((lane & 16) ? 8 : 0);

  v8f acc[WI][WJ] = {};
  v16bf aA[WI], bA[WJ], aB[WI], bB[WJ];
#pragma unroll
  for (int i = 0; i < WI; ++i) aA[i] = frag_a_at(pa[i], 0);
#pragma unroll
  for (int j = 0; j < WJ; ++j) bA[j] = frag_b_at(pb[j], 0);

  for (int k0 = 0; k0 < K; k0 += 64) {
    // odd step (k0+32): immediate offsets, always in range since K % 64 == 0
#pragma unroll
    for (int i = 0; i < WI; ++i) aB[i] = frag_a_at(pa[i], 32);
#pragma unroll
    for (int j = 0; j < WJ; ++j) bB[j] = frag_b_at(pb[j], 32);
    // consume even step
#pragma unroll
    for (int i = 0; i < WI; ++i)
#pragma unroll
      for (int j = 0; j < WJ; ++j)
        acc[i][j] = __builtin_amdgcn_wmma_f32_16x16x32_bf16(
            false, aA[i], false, bA[j], (short)0, acc[i][j], false, false);
    // advance base pointers (0 on last iteration -> dummy in-range reload)
    const int adv = (k0 + 64 < K) ? 64 : 0;
#pragma unroll
    for (int i = 0; i < WI; ++i) pa[i] += adv;
#pragma unroll
    for (int j = 0; j < WJ; ++j) pb[j] += adv;
#pragma unroll
    for (int i = 0; i < WI; ++i) aA[i] = frag_a_at(pa[i], 0);
#pragma unroll
    for (int j = 0; j < WJ; ++j) bA[j] = frag_b_at(pb[j], 0);
    // consume odd step
#pragma unroll
    for (int i = 0; i < WI; ++i)
#pragma unroll
      for (int j = 0; j < WJ; ++j)
        acc[i][j] = __builtin_amdgcn_wmma_f32_16x16x32_bf16(
            false, aB[i], false, bB[j], (short)0, acc[i][j], false, false);
  }

  const int hiOff = (lane & 16) ? 8 : 0;
#pragma unroll
  for (int wi = 0; wi < WI; ++wi) {
#pragma unroll
    for (int wj = 0; wj < WJ; ++wj) {
      const int jg = tileJ + wj * 16 + (lane & 15);
#pragma unroll
      for (int r = 0; r < 8; ++r) {
        const int ig = tileI + wi * 16 + r + hiOff;
        float v = acc[wi][wj][r];
        if constexpr (MODE == EPI_BF16) {
          outB[(size_t)z * batchO + (size_t)ig * ldo + jg] = (__bf16)v;
        } else if constexpr (MODE == EPI_F32) {
          outF[(size_t)z * batchO + (size_t)ig * ldo + jg] = v;
        } else if constexpr (MODE == EPI_BIAS_BF16) {
          outB[(size_t)z * batchO + (size_t)ig * ldo + jg] = (__bf16)(v + p1[ig]);
        } else if constexpr (MODE == EPI_SUBX_T_BF16) {
          // t = x - x_r, stored transposed: tT[j][i]
          float xv = p0[(size_t)z * p0b + (size_t)ig * ldp + jg];
          outB[(size_t)z * batchO + (size_t)jg * ldo + ig] = (__bf16)(xv - v);
        } else if constexpr (MODE == EPI_HEAD) {
          // rows i = position m, cols j = channel e; y = BN(acc + b_t), out = x + relu(y)
          const int e = jg, m = ig;
          float val   = v + p1[e];
          float scale = p2[e] * rsqrtf(p5[e] + 1e-5f);
          float y     = (val - p4[e]) * scale + p3[e];
          y           = fmaxf(y, 0.0f);
          float xres  = p0[(size_t)z * p0b + (size_t)e * ldp + m];
          outB[(size_t)z * batchO + (size_t)m * ldo + (coff + e)] = (__bf16)(xres + y);
        } else if constexpr (MODE == EPI_FINAL) {
          const int c = ig;
          float y = (v - p4[c]) * (p2[c] * rsqrtf(p5[c] + 1e-5f)) + p3[c];
          outF[(size_t)z * batchO + (size_t)c * ldo + jg] = (y >= 0.0f) ? y : 0.2f * y;
        }
      }
    }
  }
}

__global__ __launch_bounds__(256) void cvt_bf16_kernel(const float* __restrict__ in,
                                                       __bf16* __restrict__ out, int n) {
  int i = blockIdx.x * 256 + threadIdx.x;
  if (i < n) out[i] = (__bf16)in[i];
}

// [B][rows][cols] f32 -> [B][cols][rows] bf16 via LDS 32x32 tiles
__global__ __launch_bounds__(256) void transpose_cvt_kernel(const float* __restrict__ in,
                                                            __bf16* __restrict__ out,
                                                            int rows, int cols) {
  __shared__ float tile[32][33];
  const int b = blockIdx.z;
  const int r0 = blockIdx.y * 32, c0 = blockIdx.x * 32;
  const int tx = threadIdx.x & 31, ty = threadIdx.x >> 5;
  const float* src = in + (size_t)b * rows * cols;
  for (int i = ty; i < 32; i += 8) tile[i][tx] = src[(size_t)(r0 + i) * cols + c0 + tx];
  __syncthreads();
  __bf16* dst = out + (size_t)b * rows * cols;
  for (int i = ty; i < 32; i += 8)
    dst[(size_t)(c0 + i) * rows + r0 + tx] = (__bf16)tile[tx][i];
}

// in-place row softmax on S [B][n][n]; one block per row
__global__ __launch_bounds__(256) void softmax_row_kernel(float* __restrict__ S, int n) {
  __shared__ float red[256];
  float* p = S + ((size_t)blockIdx.y * n + blockIdx.x) * n;
  const int t = threadIdx.x;
  float mx = -3.4e38f;
  for (int j = t; j < n; j += 256) mx = fmaxf(mx, p[j]);
  red[t] = mx; __syncthreads();
  for (int s = 128; s > 0; s >>= 1) { if (t < s) red[t] = fmaxf(red[t], red[t + s]); __syncthreads(); }
  mx = red[0]; __syncthreads();
  float sum = 0.0f;
  for (int j = t; j < n; j += 256) { float e = __expf(p[j] - mx); p[j] = e; sum += e; }
  red[t] = sum; __syncthreads();
  for (int s = 128; s > 0; s >>= 1) { if (t < s) red[t] += red[t + s]; __syncthreads(); }
  const float inv = 1.0f / red[0];
  for (int j = t; j < n; j += 256) p[j] *= inv;
}

__global__ __launch_bounds__(256) void colsum_kernel(const float* __restrict__ S,
                                                     float* __restrict__ cs, int n) {
  const int m = blockIdx.x * 256 + threadIdx.x, b = blockIdx.y;
  const float* p = S + (size_t)b * n * n + m;
  float s = 0.0f;
  for (int i = 0; i < n; ++i) s += p[(size_t)i * n];
  cs[b * n + m] = s;
}

// attnT[m][n] = S[n][m] / (1e-9 + colsum[m]), bf16, via LDS transpose
__global__ __launch_bounds__(256) void renorm_transpose_kernel(const float* __restrict__ S,
                                                               const float* __restrict__ cs,
                                                               __bf16* __restrict__ attnT, int n) {
  __shared__ float tile[32][33];
  const int b = blockIdx.z;
  const int n0 = blockIdx.x * 32, m0 = blockIdx.y * 32;
  const int tx = threadIdx.x & 31, ty = threadIdx.x >> 5;
  const float* src = S + (size_t)b * n * n;
  for (int i = ty; i < 32; i += 8) tile[i][tx] = src[(size_t)(n0 + i) * n + m0 + tx];
  __syncthreads();
  __bf16* dst = attnT + (size_t)b * n * n;
  for (int i = ty; i < 32; i += 8) {
    float d = 1.0f / (1e-9f + cs[b * n + m0 + i]);
    dst[(size_t)(m0 + i) * n + n0 + tx] = (__bf16)(tile[tx][i] * d);
  }
}

extern "C" void kernel_launch(void* const* d_in, const int* in_sizes, int n_in,
                              void* d_out, int out_size, void* d_ws, size_t ws_size,
                              hipStream_t stream) {
  (void)in_sizes; (void)n_in; (void)out_size; (void)ws_size;
  const float* x     = (const float*)d_in[0];
  const float* w_qk  = (const float*)d_in[1];
  const float* w_v   = (const float*)d_in[2];
  const float* b_v   = (const float*)d_in[3];
  const float* w_t   = (const float*)d_in[4];
  const float* b_t   = (const float*)d_in[5];
  const float* bn_g  = (const float*)d_in[6];
  const float* bn_b  = (const float*)d_in[7];
  const float* bn_m  = (const float*)d_in[8];
  const float* bn_v  = (const float*)d_in[9];
  const float* w_lin = (const float*)d_in[10];
  const float* lbn_g = (const float*)d_in[11];
  const float* lbn_b = (const float*)d_in[12];
  const float* lbn_m = (const float*)d_in[13];
  const float* lbn_v = (const float*)d_in[14];
  float* out = (float*)d_out;

  char* ws = (char*)d_ws;
  size_t off = 0;
  auto take = [&](size_t bytes) -> void* {
    size_t cur = (off + 255) & ~(size_t)255;
    off = cur + bytes;
    return (void*)(ws + cur);
  };

  __bf16* wqk_bf  = (__bf16*)take((size_t)HH * DD * CC * 2);
  __bf16* wv_bf   = (__bf16*)take((size_t)HH * CC * CC * 2);
  __bf16* wt_bf   = (__bf16*)take((size_t)HH * CC * CC * 2);
  __bf16* wlin_bf = (__bf16*)take((size_t)CC * HC * 2);
  __bf16* xT      = (__bf16*)take((size_t)BB * NN * CC * 2);   // [B][N][C]
  __bf16* P       = (__bf16*)take((size_t)BB * NN * DD * 2);   // [B][N][D]
  __bf16* xv      = (__bf16*)take((size_t)BB * CC * NN * 2);   // [B][C][N]
  float*  energy  = (float*) take((size_t)BB * NN * NN * 4);   // [B][N][N]
  float*  colsum  = (float*) take((size_t)BB * NN * 4);
  __bf16* attnT   = (__bf16*)take((size_t)BB * NN * NN * 2);   // [B][m][n]
  __bf16* tT      = (__bf16*)take((size_t)BB * NN * CC * 2);   // [B][N][C]
  __bf16* catT    = (__bf16*)take((size_t)BB * NN * HC * 2);   // [B][N][H*C]

  // weight conversions to bf16
  cvt_bf16_kernel<<<(HH * DD * CC + 255) / 256, 256, 0, stream>>>(w_qk, wqk_bf, HH * DD * CC);
  cvt_bf16_kernel<<<(HH * CC * CC + 255) / 256, 256, 0, stream>>>(w_v, wv_bf, HH * CC * CC);
  cvt_bf16_kernel<<<(HH * CC * CC + 255) / 256, 256, 0, stream>>>(w_t, wt_bf, HH * CC * CC);
  cvt_bf16_kernel<<<(CC * HC + 255) / 256, 256, 0, stream>>>(w_lin, wlin_bf, CC * HC);
  // xT = transpose(x) as bf16
  transpose_cvt_kernel<<<dim3(NN / 32, CC / 32, BB), 256, 0, stream>>>(x, xT, CC, NN);

  for (int h = 0; h < HH; ++h) {
    // P[n][d] = sum_c xT[n][c] * w_qk[h][d][c]   (Nd=64 -> 2x1 wave tile, block 64x64)
    wmma_gemm<EPI_BF16, 2, 1><<<dim3(DD / 64, NN / 64, BB), 256, 0, stream>>>(
        xT, wqk_bf + (size_t)h * DD * CC, CC, CC, CC,
        (long long)NN * CC, 0,
        nullptr, P, DD, (long long)NN * DD,
        nullptr, 0, 0, nullptr, nullptr, nullptr, nullptr, nullptr, 0);
    // xv[e][n] = sum_c w_v[h][e][c] * xT[n][c] + b_v[e]
    wmma_gemm<EPI_BIAS_BF16, 2, 2><<<dim3(NN / 128, CC / 64, BB), 256, 0, stream>>>(
        wv_bf + (size_t)h * CC * CC, xT, CC, CC, CC,
        0, (long long)NN * CC,
        nullptr, xv, NN, (long long)CC * NN,
        nullptr, 0, 0, b_v + h * CC, nullptr, nullptr, nullptr, nullptr, 0);
    // energy[n][m] = sum_d P[n][d] * P[m][d]   (q/k share weights -> P P^T)
    wmma_gemm<EPI_F32, 2, 2><<<dim3(NN / 128, NN / 64, BB), 256, 0, stream>>>(
        P, P, DD, DD, DD,
        (long long)NN * DD, (long long)NN * DD,
        energy, nullptr, NN, (long long)NN * NN,
        nullptr, 0, 0, nullptr, nullptr, nullptr, nullptr, nullptr, 0);
    softmax_row_kernel<<<dim3(NN, BB), 256, 0, stream>>>(energy, NN);
    colsum_kernel<<<dim3(NN / 256, BB), 256, 0, stream>>>(energy, colsum, NN);
    renorm_transpose_kernel<<<dim3(NN / 32, NN / 32, BB), 256, 0, stream>>>(energy, colsum, attnT, NN);
    // x_r[e][m] = sum_n xv[e][n] * attnT[m][n];  tT[m][e] = x[e][m] - x_r[e][m]
    wmma_gemm<EPI_SUBX_T_BF16, 2, 2><<<dim3(NN / 128, CC / 64, BB), 256, 0, stream>>>(
        xv, attnT, NN, NN, NN,
        (long long)CC * NN, (long long)NN * NN,
        nullptr, tT, CC, (long long)NN * CC,
        x, (long long)CC * NN, NN, nullptr, nullptr, nullptr, nullptr, nullptr, 0);
    // yT[m][e] = sum_c tT[m][c] * w_t[h][e][c]; catT[m][h*C+e] = x[e][m] + relu(BN(y + b_t))
    wmma_gemm<EPI_HEAD, 2, 2><<<dim3(CC / 128, NN / 64, BB), 256, 0, stream>>>(
        tT, wt_bf + (size_t)h * CC * CC, CC, CC, CC,
        (long long)NN * CC, 0,
        nullptr, catT, HC, (long long)NN * HC,
        x, (long long)CC * NN, NN,
        b_t + h * CC, bn_g + h * CC, bn_b + h * CC, bn_m + h * CC, bn_v + h * CC, h * CC);
  }
  // out[c][n] = LeakyReLU(BN(sum_k w_lin[c][k] * catT[n][k]))
  wmma_gemm<EPI_FINAL, 2, 2><<<dim3(NN / 128, CC / 64, BB), 256, 0, stream>>>(
      wlin_bf, catT, HC, HC, HC,
      0, (long long)NN * HC,
      out, nullptr, NN, (long long)CC * NN,
      nullptr, 0, 0, nullptr, lbn_g, lbn_b, lbn_m, lbn_v, 0);
}
